// MHSA_77756087927530
// MI455X (gfx1250) — matrix operand order repaired and checked by reading, not compiled, not run on hardware
//
#include <hip/hip_runtime.h>
#include <hip/hip_bf16.h>

typedef __attribute__((ext_vector_type(16))) _Float16 v16h_t;
typedef __attribute__((ext_vector_type(8)))  _Float16 v8h_t;
typedef __attribute__((ext_vector_type(8)))  float    v8f_t;

#define BB 8
#define TT 1024
#define CC 1024
#define HH 16
#define DD 64
#define BT (BB*TT)

#define NEG_INF (-__builtin_inff())

__device__ __forceinline__ v8f_t wmma_f16(v16h_t a, v16h_t b, v8f_t c) {
  // D = A(16x32 f16) x B(32x16 f16) + C(16x16 f32)
  return __builtin_amdgcn_wmma_f32_16x16x32_f16(
      /*neg_a=*/false, a, /*neg_b=*/false, b,
      /*c_mod=*/(short)0, c, /*reuse_a=*/false, /*reuse_b=*/false);
}

// ---------------------------------------------------------------------------
// CDNA5 async copy: GLOBAL_LOAD_ASYNC_TO_LDS_B128 (ASYNCcnt-tracked), GV mode.
// vdst = LDS byte address, vaddr = 64-bit global address, saddr = off.
// ---------------------------------------------------------------------------
__device__ __forceinline__ void async_ld_b128(void* lds_ptr, const void* gptr) {
  unsigned loff = (unsigned)(unsigned long long)lds_ptr;  // low 32b = LDS offset
  asm volatile("global_load_async_to_lds_b128 %0, %1, off"
               :: "v"(loff), "v"(gptr)
               : "memory");
}
__device__ __forceinline__ void wait_async0() { asm volatile("s_wait_asynccnt 0x0" ::: "memory"); }
__device__ __forceinline__ void wait_async2() { asm volatile("s_wait_asynccnt 0x2" ::: "memory"); }
__device__ __forceinline__ void wait_async4() { asm volatile("s_wait_asynccnt 0x4" ::: "memory"); }

// ---------------------------------------------------------------------------
// 1) Deterministic |W| partial sums: 4 weights x 64 blocks
// ---------------------------------------------------------------------------
__global__ void k_abs_partial(const float* __restrict__ W0, const float* __restrict__ W1,
                              const float* __restrict__ W2, const float* __restrict__ W3,
                              float* __restrict__ partial) {
  int w = blockIdx.x >> 6;
  int blk = blockIdx.x & 63;
  const float* W = (w == 0) ? W0 : (w == 1) ? W1 : (w == 2) ? W2 : W3;
  const int N = CC * CC;
  float s = 0.f;
  for (int i = blk * 256 + threadIdx.x; i < N; i += 64 * 256) s += fabsf(W[i]);
  __shared__ float red[256];
  red[threadIdx.x] = s;
  __syncthreads();
  for (int o = 128; o > 0; o >>= 1) {
    if ((int)threadIdx.x < o) red[threadIdx.x] += red[threadIdx.x + o];
    __syncthreads();
  }
  if (threadIdx.x == 0) partial[w * 64 + blk] = red[0];
}

// ---------------------------------------------------------------------------
// 2) Finalize scales: scales[w*2]=s, scales[w*2+1]=1/s
// ---------------------------------------------------------------------------
__global__ void k_scales(const float* __restrict__ partial, float* __restrict__ scales,
                         const int* __restrict__ bw) {
  int w = threadIdx.x;
  if (w < 4) {
    float sum = 0.f;
    for (int i = 0; i < 64; i++) sum += partial[w * 64 + i];
    float mean = sum * (1.0f / (float)(CC * CC));
    float Qp = (float)((1 << (bw[0] - 1)) - 1);
    float s = Qp / fmaxf(mean, 1e-5f);
    scales[w * 2 + 0] = s;
    scales[w * 2 + 1] = 1.0f / s;
  }
}

// ---------------------------------------------------------------------------
// 3) Quantize weights to ternary (general Qp), stored f16 row-major [o][c]
// ---------------------------------------------------------------------------
__global__ void k_quant(const float* __restrict__ Wq, const float* __restrict__ Wk,
                        const float* __restrict__ Wv, const float* __restrict__ Wo,
                        _Float16* __restrict__ Tq, _Float16* __restrict__ Tk,
                        _Float16* __restrict__ Tv, _Float16* __restrict__ To,
                        const float* __restrict__ scales, const int* __restrict__ bw) {
  int w = blockIdx.y;
  const float* W = (w == 0) ? Wq : (w == 1) ? Wk : (w == 2) ? Wv : Wo;
  _Float16* Tt = (w == 0) ? Tq : (w == 1) ? Tk : (w == 2) ? Tv : To;
  float s = scales[w * 2];
  float Qp = (float)((1 << (bw[0] - 1)) - 1);
  int i = blockIdx.x * 256 + threadIdx.x;
  Tt[i] = (_Float16)fminf(fmaxf(rintf(W[i] * s), -Qp), Qp);
}

// ---------------------------------------------------------------------------
// 4) LayerNorm: one block (256 thr) per row, output f16
// ---------------------------------------------------------------------------
__global__ void k_layernorm(const float* __restrict__ x, const float* __restrict__ gamma,
                            const float* __restrict__ beta, _Float16* __restrict__ y) {
  int row = blockIdx.x;
  const float* xr = x + (size_t)row * CC;
  float v[4];
  float s = 0.f;
#pragma unroll
  for (int j = 0; j < 4; j++) {
    v[j] = xr[threadIdx.x + j * 256];
    s += v[j];
  }
  __shared__ float red[256];
  red[threadIdx.x] = s;
  __syncthreads();
  for (int o = 128; o > 0; o >>= 1) {
    if ((int)threadIdx.x < o) red[threadIdx.x] += red[threadIdx.x + o];
    __syncthreads();
  }
  float mu = red[0] * (1.0f / CC);
  __syncthreads();
  float q = 0.f;
#pragma unroll
  for (int j = 0; j < 4; j++) {
    float d = v[j] - mu;
    q += d * d;
  }
  red[threadIdx.x] = q;
  __syncthreads();
  for (int o = 128; o > 0; o >>= 1) {
    if ((int)threadIdx.x < o) red[threadIdx.x] += red[threadIdx.x + o];
    __syncthreads();
  }
  float rstd = rsqrtf(red[0] * (1.0f / CC) + 1e-5f);
#pragma unroll
  for (int j = 0; j < 4; j++) {
    int col = threadIdx.x + j * 256;
    y[(size_t)row * CC + col] = (_Float16)((v[j] - mu) * rstd * gamma[col] + beta[col]);
  }
}

// ---------------------------------------------------------------------------
// Fragment loaders (wave32), contiguous k = 16*half + s order used by both A and B
//  A 16x32 f16: lane<16 row=l15 holds K {0..15}; lane>=16 holds {16..31}
//  B 32x16 f16: lane col=l15, 16 contiguous K starting at (half?16:0)
// ---------------------------------------------------------------------------
__device__ __forceinline__ v16h_t load_afrag(const _Float16* rowptr, int half) {
  const _Float16* ap = rowptr + half * 16;
  v8h_t lo = *(const v8h_t*)ap;
  v8h_t hi = *(const v8h_t*)(ap + 8);
  v16h_t a;
#pragma unroll
  for (int i = 0; i < 8; i++) { a[i] = lo[i]; a[8 + i] = hi[i]; }
  return a;
}

// 16 contiguous f16 (two 16B chunks) -> B fragment halves 0..15
__device__ __forceinline__ v16h_t lds_frag16(const unsigned char* p) {
  v8h_t lo = *(const v8h_t*)p;
  v8h_t hi = *(const v8h_t*)(p + 16);
  v16h_t b;
#pragma unroll
  for (int i = 0; i < 8; i++) { b[i] = lo[i]; b[8 + i] = hi[i]; }
  return b;
}

// ---------------------------------------------------------------------------
// 5) QKV GEMM: 256 thr / 8 waves; block tile 128(M)x128(N); K-step 32.
//    Weight K-slab [128n x 32k] double-buffered in LDS via async-to-LDS,
//    shared by all 8 waves. Rows padded 64B->80B against bank conflicts.
//    All 8 B-fragments loaded before the 8 WMMAs (one dscnt wait per K-step,
//    back-to-back XDL issue). grid = (BT/128, C/128, 3)
// ---------------------------------------------------------------------------
__global__ void __launch_bounds__(256) k_qkv(
    const _Float16* __restrict__ y, const _Float16* __restrict__ WtQ,
    const _Float16* __restrict__ WtK, const _Float16* __restrict__ WtV,
    const float* __restrict__ bq, const float* __restrict__ bk,
    const float* __restrict__ bv, const float* __restrict__ scales,
    _Float16* __restrict__ Qb, _Float16* __restrict__ Kb, _Float16* __restrict__ Vtb) {
  const int mode = blockIdx.z;
  const _Float16* Wt = (mode == 0) ? WtQ : (mode == 1) ? WtK : WtV;
  const float* bias = (mode == 0) ? bq : (mode == 1) ? bk : bv;
  const float inv_s = scales[mode * 2 + 1];
  const int tid = threadIdx.x;
  const int wid = tid >> 5;
  const int lane = tid & 31;
  const int half = lane >> 4, l15 = lane & 15;
  const int m0 = blockIdx.x * 128 + wid * 16;
  const int n0 = blockIdx.y * 128;

  __shared__ __align__(16) unsigned char Bsh[2][128 * 80];  // 20 KB

  // prologue: slab k0=0 -> buffer 0 (2 async b128 per thread)
#pragma unroll
  for (int u = 0; u < 2; u++) {
    int j = tid * 2 + u, row = j >> 2, c = j & 3;
    async_ld_b128(&Bsh[0][row * 80 + c * 16], Wt + (size_t)(n0 + row) * CC + c * 8);
  }

  v8f_t acc[8] = {};
  int buf = 0;
  for (int k0 = 0; k0 < CC; k0 += 32) {
    if (k0 + 32 < CC) {
#pragma unroll
      for (int u = 0; u < 2; u++) {
        int j = tid * 2 + u, row = j >> 2, c = j & 3;
        async_ld_b128(&Bsh[buf ^ 1][row * 80 + c * 16],
                      Wt + (size_t)(n0 + row) * CC + (k0 + 32) + c * 8);
      }
      wait_async2();  // slab k0 complete (async loads finish in order)
    } else {
      wait_async0();
    }
    __syncthreads();

    v16h_t a = load_afrag(y + (size_t)(m0 + l15) * CC + k0, half);
    const unsigned char* bs = Bsh[buf];
    v16h_t bfr[8];
#pragma unroll
    for (int nt = 0; nt < 8; nt++)
      bfr[nt] = lds_frag16(bs + (nt * 16 + l15) * 80 + half * 32);
#pragma unroll
    for (int nt = 0; nt < 8; nt++) acc[nt] = wmma_f16(a, bfr[nt], acc[nt]);
    __syncthreads();
    buf ^= 1;
  }

#pragma unroll
  for (int nt = 0; nt < 8; nt++) {
    int n = n0 + nt * 16 + l15;
    int h = n >> 6, d = n & 63;
    float bn = bias[n];
#pragma unroll
    for (int r = 0; r < 8; r++) {
      int row = m0 + r + half * 8;
      int bi = row >> 10, t = row & 1023;
      _Float16 hv = (_Float16)(acc[nt][r] * inv_s + bn);
      size_t bh = (size_t)bi * HH + h;
      if (mode == 0)
        Qb[(bh * TT + t) * DD + d] = hv;
      else if (mode == 1)
        Kb[(bh * TT + t) * DD + d] = hv;
      else
        Vtb[(bh * DD + d) * TT + t] = hv;
    }
  }
}

// ---------------------------------------------------------------------------
// 6) Flash attention per (b,h): 4 waves/block, each wave a 16-row t-tile.
//    K-chunk [32s x 64d] and Vt-chunk [64d x 32s] double-buffered in LDS via
//    async-to-LDS, shared by the 4 waves. P re-layout via per-wave LDS slice.
//    Fragments batched before WMMAs.
// ---------------------------------------------------------------------------
__global__ void __launch_bounds__(128) k_attn(
    const _Float16* __restrict__ Qb, const _Float16* __restrict__ Kb,
    const _Float16* __restrict__ Vtb, const int* __restrict__ mask,
    _Float16* __restrict__ Hhb) {
  const int bz = blockIdx.z, hy = blockIdx.y;
  const int tid = threadIdx.x;
  const int wid = tid >> 5;
  const int lane = tid & 31;
  const int half = lane >> 4, l15 = lane & 15;
  const int t0 = (blockIdx.x * 4 + wid) * 16;
  const size_t bh = (size_t)bz * HH + hy;
  const float SC = 0.125f;  // 1/sqrt(64)

  __shared__ __align__(16) unsigned char Ksh[2][32 * 144];  // rows padded 128B->144B
  __shared__ __align__(16) unsigned char Vsh[2][64 * 80];   // rows padded 64B->80B
  __shared__ __align__(32) _Float16 pbuf[4][16 * 32];
  _Float16* pb = pbuf[wid];

  auto issue_chunk = [&](int nb, int s0n) {
#pragma unroll
    for (int u = 0; u < 2; u++) {  // K: 32 rows x 8 x 16B
      int j = tid * 2 + u, row = j >> 3, c = j & 7;
      async_ld_b128(&Ksh[nb][row * 144 + c * 16],
                    Kb + (bh * TT + (size_t)(s0n + row)) * DD + c * 8);
    }
#pragma unroll
    for (int u = 0; u < 2; u++) {  // V: 64 rows x 4 x 16B
      int j = tid * 2 + u, row = j >> 2, c = j & 3;
      async_ld_b128(&Vsh[nb][row * 80 + c * 16],
                    Vtb + (bh * DD + (size_t)row) * TT + s0n + c * 8);
    }
  };

  // Q A-fragments for the two K-chunks of D=64 (read once, kept in VGPRs)
  v16h_t qa[2];
  const _Float16* qrow = Qb + (bh * TT + (size_t)(t0 + l15)) * DD;
#pragma unroll
  for (int c = 0; c < 2; c++) qa[c] = load_afrag(qrow + c * 32, half);

  float mrun[8], lrun[8];
#pragma unroll
  for (int r = 0; r < 8; r++) { mrun[r] = NEG_INF; lrun[r] = 0.f; }
  v8f_t oacc[4] = {};

  issue_chunk(0, 0);
  int buf = 0;
  for (int s0 = 0; s0 < TT; s0 += 32) {
    if (s0 + 32 < TT) {
      issue_chunk(buf ^ 1, s0 + 32);
      wait_async4();  // current chunk's 4 ops complete
    } else {
      wait_async0();
    }
    __syncthreads();

    // --- scores: two 16x16 tiles over s; all 4 K fragments first ---
    const unsigned char* ks = Ksh[buf];
    v16h_t kf[2][2];
#pragma unroll
    for (int st = 0; st < 2; st++)
#pragma unroll
      for (int c = 0; c < 2; c++)
        kf[st][c] = lds_frag16(ks + (st * 16 + l15) * 144 + c * 64 + half * 32);

    v8f_t sc[2];
#pragma unroll
    for (int st = 0; st < 2; st++) {
      v8f_t a = {};
      a = wmma_f16(qa[0], kf[st][0], a);
      a = wmma_f16(qa[1], kf[st][1], a);
      sc[st] = a;
    }

    // --- scale, mask, row-max across the 16 lanes of each half ---
    float tmax[8];
#pragma unroll
    for (int r = 0; r < 8; r++) {
      int tg = t0 + r + half * 8;
      const int* mrow = mask + ((size_t)bz * TT + tg) * TT + s0;
      float a0 = sc[0][r] * SC;
      float a1 = sc[1][r] * SC;
      if (mrow[l15] == 0) a0 = NEG_INF;
      if (mrow[16 + l15] == 0) a1 = NEG_INF;
      sc[0][r] = a0;
      sc[1][r] = a1;
      float m = fmaxf(a0, a1);
#pragma unroll
      for (int off = 1; off < 16; off <<= 1) m = fmaxf(m, __shfl_xor(m, off, 32));
      tmax[r] = m;
    }

    // --- online softmax update, stage P (f16) in per-wave LDS slice ---
#pragma unroll
    for (int r = 0; r < 8; r++) {
      float mnew = fmaxf(mrun[r], tmax[r]);
      float corr = (mrun[r] == NEG_INF) ? 0.f : expf(mrun[r] - mnew);
      float p0 = (sc[0][r] == NEG_INF) ? 0.f : expf(sc[0][r] - mnew);
      float p1 = (sc[1][r] == NEG_INF) ? 0.f : expf(sc[1][r] - mnew);
      int m = r + half * 8;
      pb[m * 32 + l15] = (_Float16)p0;
      pb[m * 32 + 16 + l15] = (_Float16)p1;
      float sres = p0 + p1;
#pragma unroll
      for (int off = 1; off < 16; off <<= 1) sres += __shfl_xor(sres, off, 32);
      lrun[r] = lrun[r] * corr + sres;
      mrun[r] = mnew;
#pragma unroll
      for (int nt = 0; nt < 4; nt++) oacc[nt][r] *= corr;
    }
    __syncthreads();

    // --- P A-fragment (16x32 over s) + P@V; all 4 V fragments first ---
    v16h_t pa = load_afrag(pb + l15 * 32, half);
    const unsigned char* vs = Vsh[buf];
    v16h_t vf[4];
#pragma unroll
    for (int nt = 0; nt < 4; nt++)
      vf[nt] = lds_frag16(vs + (nt * 16 + l15) * 80 + half * 32);
#pragma unroll
    for (int nt = 0; nt < 4; nt++) oacc[nt] = wmma_f16(pa, vf[nt], oacc[nt]);
    __syncthreads();
    buf ^= 1;
  }

  // --- normalize rows (fully-masked rows -> 0, matching nan_to_num) & store ---
#pragma unroll
  for (int r = 0; r < 8; r++) {
    float rcp = (lrun[r] > 0.f) ? (1.0f / lrun[r]) : 0.f;
    int tg = t0 + r + half * 8;
#pragma unroll
    for (int nt = 0; nt < 4; nt++) {
      int col = hy * DD + nt * 16 + l15;
      Hhb[((size_t)bz * TT + tg) * CC + col] = (_Float16)(oacc[nt][r] * rcp);
    }
  }
}

// ---------------------------------------------------------------------------
// 7) Output projection: Hh @ Wo^T *inv_s + bo, * seq_mask, + x -> fp32 out.
//    Same 128x128 block tile + async LDS weight staging as k_qkv.
// ---------------------------------------------------------------------------
__global__ void __launch_bounds__(256) k_oproj(
    const _Float16* __restrict__ Hhb, const _Float16* __restrict__ WtO,
    const float* __restrict__ bo, const float* __restrict__ scales,
    const float* __restrict__ x, const int* __restrict__ mask,
    float* __restrict__ out) {
  const float inv_s = scales[3 * 2 + 1];
  const int tid = threadIdx.x;
  const int wid = tid >> 5;
  const int lane = tid & 31;
  const int half = lane >> 4, l15 = lane & 15;
  const int m0 = blockIdx.x * 128 + wid * 16;
  const int n0 = blockIdx.y * 128;

  __shared__ __align__(16) unsigned char Bsh[2][128 * 80];

#pragma unroll
  for (int u = 0; u < 2; u++) {
    int j = tid * 2 + u, row = j >> 2, c = j & 3;
    async_ld_b128(&Bsh[0][row * 80 + c * 16], WtO + (size_t)(n0 + row) * CC + c * 8);
  }

  v8f_t acc[8] = {};
  int buf = 0;
  for (int k0 = 0; k0 < CC; k0 += 32) {
    if (k0 + 32 < CC) {
#pragma unroll
      for (int u = 0; u < 2; u++) {
        int j = tid * 2 + u, row = j >> 2, c = j & 3;
        async_ld_b128(&Bsh[buf ^ 1][row * 80 + c * 16],
                      WtO + (size_t)(n0 + row) * CC + (k0 + 32) + c * 8);
      }
      wait_async2();
    } else {
      wait_async0();
    }
    __syncthreads();

    v16h_t a = load_afrag(Hhb + (size_t)(m0 + l15) * CC + k0, half);
    const unsigned char* bs = Bsh[buf];
    v16h_t bfr[8];
#pragma unroll
    for (int nt = 0; nt < 8; nt++)
      bfr[nt] = lds_frag16(bs + (nt * 16 + l15) * 80 + half * 32);
#pragma unroll
    for (int nt = 0; nt < 8; nt++) acc[nt] = wmma_f16(a, bfr[nt], acc[nt]);
    __syncthreads();
    buf ^= 1;
  }

#pragma unroll
  for (int nt = 0; nt < 8; nt++) {
    int n = n0 + nt * 16 + l15;
    float bn = bo[n];
#pragma unroll
    for (int r = 0; r < 8; r++) {
      int row = m0 + r + half * 8;
      int bi = row >> 10, t = row & 1023;
      float sm = (mask[((size_t)bi * TT + t) * TT] != 0) ? 1.f : 0.f;
      float val = (acc[nt][r] * inv_s + bn) * sm;
      out[(size_t)row * CC + n] = x[(size_t)row * CC + n] + val;
    }
  }
}

// ---------------------------------------------------------------------------
extern "C" void kernel_launch(void* const* d_in, const int* in_sizes, int n_in,
                              void* d_out, int out_size, void* d_ws, size_t ws_size,
                              hipStream_t stream) {
  const float* x = (const float*)d_in[0];
  const int* mask = (const int*)d_in[1];
  const float* gamma = (const float*)d_in[2];
  const float* beta = (const float*)d_in[3];
  const float* Wq = (const float*)d_in[4];
  const float* bq = (const float*)d_in[5];
  const float* Wk = (const float*)d_in[6];
  const float* bk = (const float*)d_in[7];
  const float* Wv = (const float*)d_in[8];
  const float* bv = (const float*)d_in[9];
  const float* Wo = (const float*)d_in[10];
  const float* bo = (const float*)d_in[11];
  const int* bw = (const int*)d_in[12];
  float* out = (float*)d_out;

  char* wsb = (char*)d_ws;
  float* partial = (float*)wsb;          // 256 floats
  float* scales = (float*)(wsb + 2048);  // 8 floats
  size_t off = 4096;
  _Float16* WtQ = (_Float16*)(wsb + off); off += (size_t)CC * CC * 2;
  _Float16* WtK = (_Float16*)(wsb + off); off += (size_t)CC * CC * 2;
  _Float16* WtV = (_Float16*)(wsb + off); off += (size_t)CC * CC * 2;
  _Float16* WtO = (_Float16*)(wsb + off); off += (size_t)CC * CC * 2;
  _Float16* yb  = (_Float16*)(wsb + off); off += (size_t)BT * CC * 2;
  _Float16* Qb  = (_Float16*)(wsb + off); off += (size_t)BT * CC * 2;
  _Float16* Kb  = (_Float16*)(wsb + off); off += (size_t)BT * CC * 2;
  _Float16* Vtb = (_Float16*)(wsb + off); off += (size_t)BT * CC * 2;
  _Float16* Hhb = (_Float16*)(wsb + off); off += (size_t)BT * CC * 2;

  k_abs_partial<<<dim3(256), dim3(256), 0, stream>>>(Wq, Wk, Wv, Wo, partial);
  k_scales<<<dim3(1), dim3(32), 0, stream>>>(partial, scales, bw);
  k_quant<<<dim3(CC * CC / 256, 4), dim3(256), 0, stream>>>(Wq, Wk, Wv, Wo, WtQ, WtK, WtV,
                                                            WtO, scales, bw);
  k_layernorm<<<dim3(BT), dim3(256), 0, stream>>>(x, gamma, beta, yb);
  k_qkv<<<dim3(BT / 128, CC / 128, 3), dim3(256), 0, stream>>>(yb, WtQ, WtK, WtV, bq, bk, bv,
                                                               scales, Qb, Kb, Vtb);
  k_attn<<<dim3(TT / 64, HH, BB), dim3(128), 0, stream>>>(Qb, Kb, Vtb, mask, Hhb);
  k_oproj<<<dim3(BT / 128, CC / 128), dim3(256), 0, stream>>>(Hhb, WtO, bo, scales, x, mask,
                                                              out);
}
